// PlenoxelModel_3985729650737
// MI455X (gfx1250) — matrix operand order repaired and checked by reading, not compiled, not run on hardware
//
#include <hip/hip_runtime.h>
#include <hip/hip_bf16.h>
#include <math.h>

typedef __attribute__((ext_vector_type(16))) _Float16 v16h;
typedef __attribute__((ext_vector_type(8)))  float    v8f;

#define VGRID 128
#define CH    28
#define WPB   4          // waves (rays) per block
#define TPB   (WPB * 32)

__global__ __launch_bounds__(TPB)
void plenoxel_render_kernel(const float* __restrict__ vox,
                            const float* __restrict__ org,
                            const float* __restrict__ dir,
                            const int*   __restrict__ nsp,
                            float* __restrict__ out,
                            int nrays)
{
    // per-wave scratch: [corner-half][sample-slot][channel]
    __shared__ __align__(16) float lvox[WPB][2][16][CH];

    const int lane = threadIdx.x & 31;
    const int wv   = threadIdx.x >> 5;
    const int rid  = blockIdx.x * WPB + wv;
    const int ray  = (rid < nrays) ? rid : (nrays - 1);
    const int N    = nsp[0];
    const int ngroups = (N + 15) >> 4;

    // ---- per-ray setup (wave-uniform; every lane computes redundantly) ----
    float ox = org[ray*3+0], oy = org[ray*3+1], oz = org[ray*3+2];
    float dx = dir[ray*3+0], dy = dir[ray*3+1], dz = dir[ray*3+2];
    float rn = rsqrtf(dx*dx + dy*dy + dz*dz);
    dx *= rn; dy *= rn; dz *= rn;

    const float Y00 = 0.28209479177387814f;
    const float C1  = 0.4886025119029199f;
    const float C2H = 1.0925484305920792f;
    const float C2Q = 0.5462742152960396f;
    const float C20 = 0.31539156525252005f;
    float ct  = fminf(fmaxf(dz, -1.f), 1.f);
    float st  = sqrtf(fmaxf(1.f - ct*ct, 0.f));
    float phi = atan2f(dy, dx);
    float cp  = cosf(phi), sp = sinf(phi);
    float sh[9];
    sh[0] = Y00;
    sh[1] = C1 * st * sp;
    sh[2] = C1 * ct;
    sh[3] = C1 * st * cp;
    sh[4] = C2H * (st*cp) * (st*sp);
    sh[5] = C2H * st * sp * ct;
    sh[6] = C20 * (3.f*ct*ct - 1.f);
    sh[7] = C2H * st * cp * ct;
    sh[8] = C2Q * ((st*cp)*(st*cp) - (st*sp)*(st*sp));

    // ---- B fragment (ray-constant): 32x16, B[k][n] = (k<27 && k/9==n) ? sh[k%9] : 0
    // Layout (dense 16-bit B, contiguous k order, same as A):
    //   lanes 0-15 : rows k = 0..15  (half h == k),      N = lane
    //   lanes 16-31: rows k = 16..31 (half h == k-16),   N = lane-16
    const int bn    = lane & 15;
    const int kbase = (lane < 16) ? 0 : 16;
    v16h bfrag;
    #pragma unroll
    for (int h = 0; h < 16; ++h) {
        int k = kbase + h;
        float v = 0.f;
        if ((k < 27) & ((k / 9) == bn)) v = sh[k % 9];
        bfrag[h] = (_Float16)v;
    }

    const int sl = lane >> 1;   // sample slot (0..15) this lane serves
    const int phh = lane & 1;   // corner half: corners phh*4 .. phh*4+3

    float acc_lo = 0.f, acc_hi = 0.f, T = 1.f;

    for (int g = 0; g < ngroups; ++g) {
        // ---------- phase 1: gather + trilinear (28 channels in regs) ----------
        int   s  = g*16 + sl;
        float t  = (float)s;                       // T_NEAR=0, STEP=1
        float px = ox + dx*t, py = oy + dy*t, pz = oz + dz*t;
        float fx = floorf(px), fy = floorf(py), fz = floorf(pz);
        int   ix = (int)fx,  iy = (int)fy,  iz = (int)fz;
        float frx = px - fx, fry = py - fy, frz = pz - fz;
        int live = (s < N) ? 1 : 0;

        float4 acc4[7];
        #pragma unroll
        for (int q = 0; q < 7; ++q) acc4[q] = make_float4(0.f, 0.f, 0.f, 0.f);

        #pragma unroll
        for (int jj = 0; jj < 4; ++jj) {
            int j   = phh*4 + jj;
            int axo = (j >> 2) & 1, ayo = (j >> 1) & 1, azo = j & 1;
            int cx = ix + axo, cy = iy + ayo, cz = iz + azo;
            int valid = live & (cx >= 0) & (cx < VGRID)
                             & (cy >= 0) & (cy < VGRID)
                             & (cz >= 0) & (cz < VGRID);
            int bx = min(max(cx, 0), VGRID-1);
            int by = min(max(cy, 0), VGRID-1);
            int bz = min(max(cz, 0), VGRID-1);
            float w = (axo ? frx : 1.f-frx) * (ayo ? fry : 1.f-fry) * (azo ? frz : 1.f-frz);
            w = valid ? w : 0.f;
            const float4* src = (const float4*)vox + (size_t)((bx*VGRID + by)*VGRID + bz) * 7;
            #pragma unroll
            for (int q = 0; q < 7; ++q) {       // 7 x global_load_b128 = 112B contiguous
                float4 d = src[q];
                acc4[q].x += w * d.x;  acc4[q].y += w * d.y;
                acc4[q].z += w * d.z;  acc4[q].w += w * d.w;
            }
        }
        float4* dst = (float4*)&lvox[wv][phh][sl][0];
        #pragma unroll
        for (int q = 0; q < 7; ++q) dst[q] = acc4[q];

        __syncthreads();

        // ---------- phase 2: SH projection via WMMA ----------
        // A fragment: 16x32 f16, M = sample slot, K = coeff index (27 used, pad 32)
        // lanes 0-15 : halves -> K = {0..15};  lanes 16-31: K = {16..31}
        const int m = lane & 15;
        v16h afrag;
        #pragma unroll
        for (int h = 0; h < 16; ++h) {
            int K = (lane < 16) ? h
                                : h + 16;
            float v = 0.f;
            if (K < 27) v = lvox[wv][0][m][K+1] + lvox[wv][1][m][K+1];
            afrag[h] = (_Float16)v;
        }

        v8f cfrag = {};
        cfrag = __builtin_amdgcn_wmma_f32_16x16x32_f16(
            /*neg_a=*/false, afrag, /*neg_b=*/false, bfrag,
            /*c_mod=*/(short)0, cfrag, /*reuse_a=*/false, /*reuse_b=*/false);
        // C layout: lanes 0-15 hold samples 0-7 (VGPR idx = sample, channel = lane)
        //           lanes 16-31 hold samples 8-15 (VGPR idx = sample-8, channel = lane-16)

        // ---------- phase 3: sequential transmittance accumulation ----------
        #pragma unroll
        for (int ss = 0; ss < 8; ++ss) {
            float sig = lvox[wv][0][ss][0] + lvox[wv][1][ss][0];
            sig = fmaxf(sig, 0.f);
            float e  = __expf(-sig);
            float wt = T * (1.f - e);
            T *= e;
            float col = fminf(fmaxf(cfrag[ss], 0.f), 1.f);
            acc_lo += wt * col;
        }
        #pragma unroll
        for (int ss = 8; ss < 16; ++ss) {
            float sig = lvox[wv][0][ss][0] + lvox[wv][1][ss][0];
            sig = fmaxf(sig, 0.f);
            float e  = __expf(-sig);
            float wt = T * (1.f - e);
            T *= e;
            float col = fminf(fmaxf(cfrag[ss-8], 0.f), 1.f);
            acc_hi += wt * col;
        }

        __syncthreads();   // before next group overwrites LDS
    }

    // combine halves: channel c lives in lane c (acc_lo) and lane 16+c (acc_hi)
    float hi = __shfl(acc_hi, (lane & 15) + 16, 32);
    if ((rid < nrays) && (lane < 3))
        out[rid*3 + lane] = acc_lo + hi + T;     // + background term exp(-csum)
}

extern "C" void kernel_launch(void* const* d_in, const int* in_sizes, int n_in,
                              void* d_out, int out_size, void* d_ws, size_t ws_size,
                              hipStream_t stream) {
    const float* vox = (const float*)d_in[0];   // [128,128,128,28] f32
    const float* org = (const float*)d_in[1];   // [2048,3] f32
    const float* dir = (const float*)d_in[2];   // [2048,3] f32
    const int*   nsp = (const int*)d_in[3];     // scalar num_samples
    float* out = (float*)d_out;                 // [2048,3] f32

    int nrays  = in_sizes[1] / 3;
    int blocks = (nrays + WPB - 1) / WPB;
    plenoxel_render_kernel<<<blocks, TPB, 0, stream>>>(vox, org, dir, nsp, out, nrays);
}
